// BrahmanTransformer_11201274708087
// MI455X (gfx1250) — compile-verified, matrix operand order checked by reading, not run on hardware
//
#include <hip/hip_runtime.h>
#include <math.h>

// ---------------------------------------------------------------------------
// Problem constants (match reference)
// ---------------------------------------------------------------------------
#define Bc 32
#define Lc 512
#define Dc 512
#define Hc 8
#define NLc 6
#define Fc 2048
#define Cc 9
#define HDc 64
#define Mc (Bc * Lc)          // 16384 tokens

// ---------------------------------------------------------------------------
// Types for WMMA
// ---------------------------------------------------------------------------
typedef __attribute__((ext_vector_type(16))) __bf16 v16bf;
typedef __attribute__((ext_vector_type(8)))  float  v8f;

union BFrag {                 // one 16x32 bf16 A/B fragment (8 VGPRs/lane)
    v16bf v;
    uint4 q[2];
};

__device__ __forceinline__ unsigned short f2bf(float f) {
    unsigned int u = __float_as_uint(f);
    unsigned int r = u + 0x7FFFu + ((u >> 16) & 1u);   // round-to-nearest-even
    return (unsigned short)(r >> 16);
}

// ---------------------------------------------------------------------------
// CDNA5 async global->LDS path (ASYNCcnt), with sync fallback.
// Builtin signature (from clang diagnostic): (int4 AS1*, int4 AS3*, Ii, Ii).
// ---------------------------------------------------------------------------
#if defined(__gfx1250__) && __has_builtin(__builtin_amdgcn_global_load_async_to_lds_b128)
#define USE_ASYNC 1
typedef int v4i_t __attribute__((vector_size(16)));
typedef __attribute__((address_space(1))) v4i_t gv4i;
typedef __attribute__((address_space(3))) v4i_t lv4i;
#define GPTR(p) ((gv4i*)(unsigned long long)(size_t)(p))
#define LPTR(p) ((lv4i*)(unsigned)(size_t)(p))
#if __has_builtin(__builtin_amdgcn_s_wait_asynccnt)
#define WAIT_ASYNC(n) __builtin_amdgcn_s_wait_asynccnt(n)
#else
#define WAIT_ASYNC(n) asm volatile("s_wait_asynccnt %0" ::"i"(n) : "memory")
#endif
#else
#define USE_ASYNC 0
#endif

// ---------------------------------------------------------------------------
// f32 [NL,K,N] -> bf16 [NL,N,K] transposed weight conversion (once per launch)
// ---------------------------------------------------------------------------
__global__ void cvt_t_kernel(const float* __restrict__ src,
                             unsigned short* __restrict__ dst,
                             int K, int N, int total) {
    int i = blockIdx.x * 256 + threadIdx.x;
    if (i >= total) return;
    const int kn = K * N;
    const int l = i / kn;
    const int r = i - l * kn;
    const int k = r / N;
    const int n = r - k * N;
    dst[(size_t)l * kn + (size_t)n * K + k] = f2bf(src[i]);
}

// ---------------------------------------------------------------------------
// LayerNorm: one block (128 threads) per token row of 512
// ---------------------------------------------------------------------------
__global__ __launch_bounds__(128)
void ln_kernel(const float* __restrict__ x, const float* __restrict__ g,
               const float* __restrict__ beta,
               unsigned short* obf, float* of32) {
    const int row = blockIdx.x;
    const int tid = threadIdx.x;
    const float4 v = ((const float4*)(x + (size_t)row * Dc))[tid];

    float s = v.x + v.y + v.z + v.w;
#pragma unroll
    for (int off = 16; off > 0; off >>= 1) s += __shfl_xor(s, off, 32);
    __shared__ float red[4];
    const int w = tid >> 5;
    if ((tid & 31) == 0) red[w] = s;
    __syncthreads();
    const float mu = (red[0] + red[1] + red[2] + red[3]) * (1.0f / (float)Dc);

    float dx0 = v.x - mu, dx1 = v.y - mu, dx2 = v.z - mu, dx3 = v.w - mu;
    float s2 = dx0 * dx0 + dx1 * dx1 + dx2 * dx2 + dx3 * dx3;
#pragma unroll
    for (int off = 16; off > 0; off >>= 1) s2 += __shfl_xor(s2, off, 32);
    __syncthreads();
    if ((tid & 31) == 0) red[w] = s2;
    __syncthreads();
    const float var = (red[0] + red[1] + red[2] + red[3]) * (1.0f / (float)Dc);
    const float rstd = rsqrtf(var + 1e-5f);

    const float dx[4] = {dx0, dx1, dx2, dx3};
#pragma unroll
    for (int e = 0; e < 4; e++) {
        const int c = tid * 4 + e;
        const float y = dx[e] * rstd * g[c] + beta[c];
        if (obf)  obf[(size_t)row * Dc + c] = f2bf(y);
        if (of32) of32[(size_t)row * Dc + c] = y;
    }
}

// ---------------------------------------------------------------------------
// bf16 GEMM with fp32 accumulate.  C[M,N] = A[M,K] @ Wt[N,K]^T + bias
// Block tile 128x128, BK=32; 8 waves in 2(M) x 4(N) grid; each wave owns a
// 64x32 tile = 8 v_wmma_f32_16x16x32_bf16 per K-step.  A and Bt tiles are
// staged with GLOBAL_LOAD_ASYNC_TO_LDS_B128 in a double-buffered pipeline
// (s_wait_asynccnt), falling back to synchronous b128 copies.
// EPI: 0 = bias -> bf16, 1 = bias + residual -> f32, 2 = bias + GELU -> bf16
// ---------------------------------------------------------------------------
template <int EPI>
__global__ __launch_bounds__(256)
void gemm_bf16(const unsigned short* __restrict__ A,    // [M,K]
               const unsigned short* __restrict__ Wt,   // [N,K] (pre-transposed)
               const float* __restrict__ bias,
               const float* resid, float* outF,
               unsigned short* outB,
               int M, int N, int K) {
#if USE_ASYNC
    __shared__ unsigned short Asl[2][128 * 32];
    __shared__ unsigned short Btl[2][128 * 32];
#else
    __shared__ unsigned short Asl[1][128 * 32];
    __shared__ unsigned short Btl[1][128 * 32];
#endif

    const int tid  = threadIdx.x;
    const int lane = tid & 31, wid = tid >> 5;
    const int wm = wid >> 2, wn = wid & 3;
    const int hi = lane >> 4, ln15 = lane & 15;
    const int m0 = blockIdx.x * 128;
    const int n0 = blockIdx.y * 128;

    // staging assignment: 2 threads per 32-elem row, 16 bf16 (32B) each
    const int srow = tid >> 1, sseg = tid & 1;
    const unsigned short* aRow = A  + (size_t)(m0 + srow) * K + sseg * 16;
    const unsigned short* bRow = Wt + (size_t)(n0 + srow) * K + sseg * 16;
    const int ldsOff = srow * 32 + sseg * 16;

    const v8f zero = {0.f, 0.f, 0.f, 0.f, 0.f, 0.f, 0.f, 0.f};
    v8f acc[4][2];
#pragma unroll
    for (int mt = 0; mt < 4; mt++)
#pragma unroll
        for (int nt = 0; nt < 2; nt++) acc[mt][nt] = zero;

    const int nk = K >> 5;

#if USE_ASYNC
    // ---- issue tile 0
    __builtin_amdgcn_global_load_async_to_lds_b128(
        GPTR(aRow), LPTR(&Asl[0][ldsOff]), 0, 0);
    __builtin_amdgcn_global_load_async_to_lds_b128(
        GPTR(aRow + 8), LPTR(&Asl[0][ldsOff + 8]), 0, 0);
    __builtin_amdgcn_global_load_async_to_lds_b128(
        GPTR(bRow), LPTR(&Btl[0][ldsOff]), 0, 0);
    __builtin_amdgcn_global_load_async_to_lds_b128(
        GPTR(bRow + 8), LPTR(&Btl[0][ldsOff + 8]), 0, 0);

    int cur = 0;
    for (int t = 0; t < nk; t++) {
        if (t + 1 < nk) {                      // prefetch tile t+1
            const int k1 = (t + 1) * 32;
            const int nxt = cur ^ 1;
            __builtin_amdgcn_global_load_async_to_lds_b128(
                GPTR(aRow + k1), LPTR(&Asl[nxt][ldsOff]), 0, 0);
            __builtin_amdgcn_global_load_async_to_lds_b128(
                GPTR(aRow + k1 + 8), LPTR(&Asl[nxt][ldsOff + 8]), 0, 0);
            __builtin_amdgcn_global_load_async_to_lds_b128(
                GPTR(bRow + k1), LPTR(&Btl[nxt][ldsOff]), 0, 0);
            __builtin_amdgcn_global_load_async_to_lds_b128(
                GPTR(bRow + k1 + 8), LPTR(&Btl[nxt][ldsOff + 8]), 0, 0);
            WAIT_ASYNC(4);                     // tile t done (this wave)
        } else {
            WAIT_ASYNC(0);
        }
        __syncthreads();                       // tile t done (all waves)

        BFrag a[4], b[2];
#pragma unroll
        for (int mt = 0; mt < 4; mt++) {
            const uint4* s = (const uint4*)(&Asl[cur][(wm * 64 + mt * 16 + ln15) * 32 + hi * 16]);
            a[mt].q[0] = s[0]; a[mt].q[1] = s[1];
        }
#pragma unroll
        for (int nt = 0; nt < 2; nt++) {
            const uint4* s = (const uint4*)(&Btl[cur][(wn * 32 + nt * 16 + ln15) * 32 + hi * 16]);
            b[nt].q[0] = s[0]; b[nt].q[1] = s[1];
        }
#pragma unroll
        for (int mt = 0; mt < 4; mt++)
#pragma unroll
            for (int nt = 0; nt < 2; nt++)
                acc[mt][nt] = __builtin_amdgcn_wmma_f32_16x16x32_bf16(
                    false, a[mt].v, false, b[nt].v, (short)0, acc[mt][nt],
                    false, false);
        __syncthreads();                       // reads done before rewrite
        cur ^= 1;
    }
#else
    for (int t = 0; t < nk; t++) {
        const int k0 = t * 32;
        {
            const uint4* sa = (const uint4*)(aRow + k0);
            uint4 a0 = sa[0], a1 = sa[1];
            uint4* da = (uint4*)(&Asl[0][ldsOff]);
            da[0] = a0; da[1] = a1;
            const uint4* sb = (const uint4*)(bRow + k0);
            uint4 b0 = sb[0], b1 = sb[1];
            uint4* db = (uint4*)(&Btl[0][ldsOff]);
            db[0] = b0; db[1] = b1;
        }
        __syncthreads();

        BFrag a[4], b[2];
#pragma unroll
        for (int mt = 0; mt < 4; mt++) {
            const uint4* s = (const uint4*)(&Asl[0][(wm * 64 + mt * 16 + ln15) * 32 + hi * 16]);
            a[mt].q[0] = s[0]; a[mt].q[1] = s[1];
        }
#pragma unroll
        for (int nt = 0; nt < 2; nt++) {
            const uint4* s = (const uint4*)(&Btl[0][(wn * 32 + nt * 16 + ln15) * 32 + hi * 16]);
            b[nt].q[0] = s[0]; b[nt].q[1] = s[1];
        }
#pragma unroll
        for (int mt = 0; mt < 4; mt++)
#pragma unroll
            for (int nt = 0; nt < 2; nt++)
                acc[mt][nt] = __builtin_amdgcn_wmma_f32_16x16x32_bf16(
                    false, a[mt].v, false, b[nt].v, (short)0, acc[mt][nt],
                    false, false);
        __syncthreads();
    }
#endif

    // --- epilogue
#pragma unroll
    for (int mt = 0; mt < 4; mt++) {
#pragma unroll
        for (int nt = 0; nt < 2; nt++) {
            const int gn = n0 + wn * 32 + nt * 16 + ln15;
            const float bval = bias[gn];
#pragma unroll
            for (int r = 0; r < 8; r++) {
                const int gm = m0 + wm * 64 + mt * 16 + r + hi * 8;
                const float val = acc[mt][nt][r] + bval;
                const size_t idx = (size_t)gm * N + gn;
                if constexpr (EPI == 0) {
                    outB[idx] = f2bf(val);
                } else if constexpr (EPI == 1) {
                    outF[idx] = resid[idx] + val;
                } else {
                    const float gel =
                        0.5f * val * (1.0f + erff(val * 0.70710678118f));
                    outB[idx] = f2bf(gel);
                }
            }
        }
    }
}

// ---------------------------------------------------------------------------
// Flash attention: one block per (b, h, 64-row q-block); 4 waves, each owns
// 16 q rows.  Causal loop over key blocks j <= qb.  Q/K frags direct from
// global (contiguous along HD); V transposed via LDS; P bounced via LDS.
// ---------------------------------------------------------------------------
__global__ __launch_bounds__(128)
void attn_kernel(const unsigned short* __restrict__ Q,
                 const unsigned short* __restrict__ Kb,
                 const unsigned short* __restrict__ V,
                 const int* __restrict__ case_ids,
                 const int* __restrict__ amask,
                 const float* __restrict__ cbias,   // [H,C,C] this layer
                 const float* __restrict__ vbias,   // [H]  this layer
                 unsigned short* __restrict__ O) {  // [B,L,D]
    const int bx = blockIdx.x;
    const int qb = bx & 7;          // q block (L/64 = 8)
    const int h  = (bx >> 3) & 7;
    const int b  = bx >> 6;

    __shared__ unsigned short Pl[64 * 64];   // P tile  [qrow][key]
    __shared__ unsigned short Vt[64 * 64];   // V tile  [hd][key]
    __shared__ int qcid[64];
    __shared__ int kcid[64];
    __shared__ float kadd[64];               // pad mask + verb bias per key

    const int tid = threadIdx.x;
    const int lane = tid & 31, w = tid >> 5;
    const int hi = lane >> 4, ln15 = lane & 15;

    if (tid < 64) qcid[tid] = case_ids[b * Lc + qb * 64 + tid];
    const float vb = vbias[h];

    // Q fragments for this wave's 16 rows (K = HD = 64 -> 2 chunks of 32)
    BFrag qa[2];
    {
        const int qrow = qb * 64 + w * 16 + ln15;
        const unsigned short* qptr =
            Q + ((size_t)(b * Lc + qrow) * Dc + h * HDc);
#pragma unroll
        for (int kc = 0; kc < 2; kc++) {
            const uint4* s = (const uint4*)(qptr + kc * 32 + hi * 16);
            qa[kc].q[0] = s[0]; qa[kc].q[1] = s[1];
        }
    }

    const v8f zero = {0.f, 0.f, 0.f, 0.f, 0.f, 0.f, 0.f, 0.f};
    float mrow[8], lrow[8];
    v8f o[4];
#pragma unroll
    for (int r = 0; r < 8; r++) { mrow[r] = -3.0e38f; lrow[r] = 0.0f; }
#pragma unroll
    for (int t = 0; t < 4; t++) o[t] = zero;

    for (int j = 0; j <= qb; j++) {
        // ---- stage per-key metadata and transposed V tile
        if (tid < 64) {
            const int kt = b * Lc + j * 64 + tid;
            const int ck = case_ids[kt];
            kcid[tid] = ck;
            kadd[tid] = (1.0f - (float)amask[kt]) * -10000.0f +
                        (ck == 8 ? vb : 0.0f);
        }
        {
            union { uint4 q[2]; unsigned short hh[16]; } p0, p1;
            const int tok = tid >> 1, ds = (tid & 1) * 32;
            const unsigned short* vp =
                V + ((size_t)(b * Lc + j * 64 + tok) * Dc + h * HDc + ds);
            p0.q[0] = ((const uint4*)vp)[0];
            p0.q[1] = ((const uint4*)vp)[1];
            p1.q[0] = ((const uint4*)(vp + 16))[0];
            p1.q[1] = ((const uint4*)(vp + 16))[1];
#pragma unroll
            for (int e = 0; e < 16; e++) Vt[(ds + e) * 64 + tok] = p0.hh[e];
#pragma unroll
            for (int e = 0; e < 16; e++) Vt[(ds + 16 + e) * 64 + tok] = p1.hh[e];
        }
        __syncthreads();

        // ---- S = (Q K^T)/8 + biases
        v8f s[4];
#pragma unroll
        for (int nt = 0; nt < 4; nt++) {
            const int key = j * 64 + nt * 16 + ln15;
            const unsigned short* kp =
                Kb + ((size_t)(b * Lc + key) * Dc + h * HDc);
            BFrag k0, k1;
            {
                const uint4* p = (const uint4*)(kp + hi * 16);
                k0.q[0] = p[0]; k0.q[1] = p[1];
            }
            {
                const uint4* p = (const uint4*)(kp + 32 + hi * 16);
                k1.q[0] = p[0]; k1.q[1] = p[1];
            }
            s[nt] = __builtin_amdgcn_wmma_f32_16x16x32_bf16(
                false, qa[0].v, false, k0.v, (short)0, zero, false, false);
            s[nt] = __builtin_amdgcn_wmma_f32_16x16x32_bf16(
                false, qa[1].v, false, k1.v, (short)0, s[nt], false, false);
        }

#pragma unroll
        for (int nt = 0; nt < 4; nt++) {
            const int col = j * 64 + nt * 16 + ln15;
            const int ck = kcid[nt * 16 + ln15];
            const float ka = kadd[nt * 16 + ln15];
#pragma unroll
            for (int r = 0; r < 8; r++) {
                const int lrow_i = w * 16 + r + hi * 8;
                const int row = qb * 64 + lrow_i;
                const int ci = qcid[lrow_i];
                float v = s[nt][r] * 0.125f;
                v += cbias[(h * Cc + ci) * Cc + ck] + ka;
                if (col > row) v += -10000.0f;
                s[nt][r] = v;
            }
        }

        // ---- flash softmax update (16-lane groups share rows)
        float mx[8];
#pragma unroll
        for (int r = 0; r < 8; r++) {
            float m = s[0][r];
            m = fmaxf(m, s[1][r]); m = fmaxf(m, s[2][r]); m = fmaxf(m, s[3][r]);
#pragma unroll
            for (int off = 1; off < 16; off <<= 1)
                m = fmaxf(m, __shfl_xor(m, off, 32));
            mx[r] = m;
        }
        float alpha[8];
#pragma unroll
        for (int r = 0; r < 8; r++) {
            const float mn = fmaxf(mrow[r], mx[r]);
            alpha[r] = __expf(mrow[r] - mn);
            mrow[r] = mn;
        }
        float rs[8];
#pragma unroll
        for (int r = 0; r < 8; r++) rs[r] = 0.0f;
#pragma unroll
        for (int nt = 0; nt < 4; nt++)
#pragma unroll
            for (int r = 0; r < 8; r++) {
                const float p = __expf(s[nt][r] - mrow[r]);
                s[nt][r] = p;
                rs[r] += p;
            }
#pragma unroll
        for (int r = 0; r < 8; r++) {
#pragma unroll
            for (int off = 1; off < 16; off <<= 1)
                rs[r] += __shfl_xor(rs[r], off, 32);
            lrow[r] = lrow[r] * alpha[r] + rs[r];
        }
#pragma unroll
        for (int t = 0; t < 4; t++)
#pragma unroll
            for (int r = 0; r < 8; r++) o[t][r] *= alpha[r];

        // ---- P -> LDS (C-layout -> A-layout bounce)
#pragma unroll
        for (int nt = 0; nt < 4; nt++)
#pragma unroll
            for (int r = 0; r < 8; r++)
                Pl[(w * 16 + r + hi * 8) * 64 + nt * 16 + ln15] =
                    f2bf(s[nt][r]);
        __syncthreads();

        // ---- O += P @ V
        BFrag pa[2];
#pragma unroll
        for (int kc = 0; kc < 2; kc++) {
            const uint4* p =
                (const uint4*)(Pl + (w * 16 + ln15) * 64 + kc * 32 + hi * 16);
            pa[kc].q[0] = p[0]; pa[kc].q[1] = p[1];
        }
#pragma unroll
        for (int hnt = 0; hnt < 4; hnt++) {
            BFrag v0, v1;
            {
                const uint4* p = (const uint4*)(Vt + (hnt * 16 + ln15) * 64 +
                                                hi * 16);
                v0.q[0] = p[0]; v0.q[1] = p[1];
            }
            {
                const uint4* p = (const uint4*)(Vt + (hnt * 16 + ln15) * 64 +
                                                32 + hi * 16);
                v1.q[0] = p[0]; v1.q[1] = p[1];
            }
            o[hnt] = __builtin_amdgcn_wmma_f32_16x16x32_bf16(
                false, pa[0].v, false, v0.v, (short)0, o[hnt], false, false);
            o[hnt] = __builtin_amdgcn_wmma_f32_16x16x32_bf16(
                false, pa[1].v, false, v1.v, (short)0, o[hnt], false, false);
        }
        __syncthreads();
    }

    // ---- normalize and write out [B,L,D]
#pragma unroll
    for (int hnt = 0; hnt < 4; hnt++)
#pragma unroll
        for (int r = 0; r < 8; r++) {
            const int row = qb * 64 + w * 16 + r + hi * 8;
            const float val = o[hnt][r] / lrow[r];
            O[(size_t)(b * Lc + row) * Dc + h * HDc + hnt * 16 + ln15] =
                f2bf(val);
        }
}

// ---------------------------------------------------------------------------
// Host side
// ---------------------------------------------------------------------------
extern "C" void kernel_launch(void* const* d_in, const int* in_sizes, int n_in,
                              void* d_out, int out_size, void* d_ws,
                              size_t ws_size, hipStream_t stream) {
    const float* x        = (const float*)d_in[0];
    const int*   case_ids = (const int*)d_in[1];
    const int*   amask    = (const int*)d_in[2];
    const float* Wq = (const float*)d_in[3];
    const float* bq = (const float*)d_in[4];
    const float* Wk = (const float*)d_in[5];
    const float* bk = (const float*)d_in[6];
    const float* Wv = (const float*)d_in[7];
    const float* bv = (const float*)d_in[8];
    const float* Wo = (const float*)d_in[9];
    const float* bo = (const float*)d_in[10];
    const float* ln1_g = (const float*)d_in[11];
    const float* ln1_b = (const float*)d_in[12];
    const float* ln2_g = (const float*)d_in[13];
    const float* ln2_b = (const float*)d_in[14];
    const float* W1 = (const float*)d_in[15];
    const float* b1 = (const float*)d_in[16];
    const float* W2 = (const float*)d_in[17];
    const float* b2 = (const float*)d_in[18];
    const float* case_bias = (const float*)d_in[19];
    const float* verb_bias = (const float*)d_in[20];
    const float* fn_g = (const float*)d_in[21];
    const float* fn_b = (const float*)d_in[22];

    size_t off = 0;
    auto take = [&](size_t bytes) -> void* {
        void* p = (char*)d_ws + off;
        off += (bytes + 255) & ~(size_t)255;
        return p;
    };

    float*          xbuf = (float*)take((size_t)Mc * Dc * sizeof(float));
    unsigned short* hbf  = (unsigned short*)take((size_t)Mc * Dc * 2);
    unsigned short* qbf  = (unsigned short*)take((size_t)Mc * Dc * 2);
    unsigned short* kbf  = (unsigned short*)take((size_t)Mc * Dc * 2);
    unsigned short* vbf  = (unsigned short*)take((size_t)Mc * Dc * 2);
    unsigned short* abf  = (unsigned short*)take((size_t)Mc * Dc * 2);
    unsigned short* fbf  = (unsigned short*)take((size_t)Mc * Fc * 2);
    unsigned short* Wqb  = (unsigned short*)take((size_t)NLc * Dc * Dc * 2);
    unsigned short* Wkb  = (unsigned short*)take((size_t)NLc * Dc * Dc * 2);
    unsigned short* Wvb  = (unsigned short*)take((size_t)NLc * Dc * Dc * 2);
    unsigned short* Wob  = (unsigned short*)take((size_t)NLc * Dc * Dc * 2);
    unsigned short* W1b  = (unsigned short*)take((size_t)NLc * Dc * Fc * 2);
    unsigned short* W2b  = (unsigned short*)take((size_t)NLc * Fc * Dc * 2);

    // weights -> bf16, pre-transposed to [NL, N, K] (once per launch)
    const int nDD = NLc * Dc * Dc;
    const int nDF = NLc * Dc * Fc;
    cvt_t_kernel<<<(nDD + 255) / 256, 256, 0, stream>>>(Wq, Wqb, Dc, Dc, nDD);
    cvt_t_kernel<<<(nDD + 255) / 256, 256, 0, stream>>>(Wk, Wkb, Dc, Dc, nDD);
    cvt_t_kernel<<<(nDD + 255) / 256, 256, 0, stream>>>(Wv, Wvb, Dc, Dc, nDD);
    cvt_t_kernel<<<(nDD + 255) / 256, 256, 0, stream>>>(Wo, Wob, Dc, Dc, nDD);
    cvt_t_kernel<<<(nDF + 255) / 256, 256, 0, stream>>>(W1, W1b, Dc, Fc, nDF);
    cvt_t_kernel<<<(nDF + 255) / 256, 256, 0, stream>>>(W2, W2b, Fc, Dc, nDF);

    // residual stream (never mutate d_in)
    (void)hipMemcpyAsync(xbuf, x, (size_t)Mc * Dc * sizeof(float),
                         hipMemcpyDeviceToDevice, stream);

    const dim3 gD(Mc / 128, Dc / 128);   // (128, 4)
    const dim3 gF(Mc / 128, Fc / 128);   // (128, 16)
    const int attnBlocks = Bc * Hc * (Lc / 64);   // 2048

    for (int i = 0; i < NLc; i++) {
        ln_kernel<<<Mc, 128, 0, stream>>>(xbuf, ln1_g + i * Dc, ln1_b + i * Dc,
                                          hbf, nullptr);
        gemm_bf16<0><<<gD, 256, 0, stream>>>(hbf, Wqb + (size_t)i * Dc * Dc,
                                             bq + i * Dc, nullptr, nullptr,
                                             qbf, Mc, Dc, Dc);
        gemm_bf16<0><<<gD, 256, 0, stream>>>(hbf, Wkb + (size_t)i * Dc * Dc,
                                             bk + i * Dc, nullptr, nullptr,
                                             kbf, Mc, Dc, Dc);
        gemm_bf16<0><<<gD, 256, 0, stream>>>(hbf, Wvb + (size_t)i * Dc * Dc,
                                             bv + i * Dc, nullptr, nullptr,
                                             vbf, Mc, Dc, Dc);
        attn_kernel<<<attnBlocks, 128, 0, stream>>>(
            qbf, kbf, vbf, case_ids, amask,
            case_bias + (size_t)i * Hc * Cc * Cc, verb_bias + i * Hc, abf);
        gemm_bf16<1><<<gD, 256, 0, stream>>>(abf, Wob + (size_t)i * Dc * Dc,
                                             bo + i * Dc, xbuf, xbuf, nullptr,
                                             Mc, Dc, Dc);
        ln_kernel<<<Mc, 128, 0, stream>>>(xbuf, ln2_g + i * Dc, ln2_b + i * Dc,
                                          hbf, nullptr);
        gemm_bf16<2><<<gF, 256, 0, stream>>>(hbf, W1b + (size_t)i * Dc * Fc,
                                             b1 + i * Fc, nullptr, nullptr,
                                             fbf, Mc, Fc, Dc);
        gemm_bf16<1><<<gD, 256, 0, stream>>>(fbf, W2b + (size_t)i * Fc * Dc,
                                             b2 + i * Dc, xbuf, xbuf, nullptr,
                                             Mc, Dc, Fc);
    }
    ln_kernel<<<Mc, 128, 0, stream>>>(xbuf, fn_g, fn_b, nullptr,
                                      (float*)d_out);
}
